// GINENoEmb_66254165508839
// MI455X (gfx1250) — matrix operand order repaired and checked by reading, not compiled, not run on hardware
//
#include <hip/hip_runtime.h>
#include <hip/hip_bf16.h>
#include <stdint.h>

typedef __attribute__((ext_vector_type(16))) _Float16 v16h;
typedef __attribute__((ext_vector_type(8)))  float    v8f;

#define NN 50000
#define EE 600000
#define CC 130
#define LL 4
#define KPAD 160      // K padded to 5 * 32
#define NPAD 144      // N padded to 9 * 16
#define NT 9          // N tiles of 16
#define KSTEPS 5      // K steps of 32
#define WPB 8         // waves per block (wave32)
#define ROWS_PER_BLOCK (WPB * 16)
#define MAT_ELEMS (NPAD * KPAD)

// ---------------- feature construction ----------------
__global__ void gine_build_features(const int* __restrict__ xt, const int* __restrict__ xk,
                                    const float* __restrict__ xs, float* __restrict__ h) {
    int i = blockIdx.x * blockDim.x + threadIdx.x;
    if (i >= NN * CC) return;
    int n = i / CC, c = i % CC;
    float v;
    if (c < 32) {
        v = (xt[n] == c) ? 1.f : 0.f;
    } else if (c < 128) {
        int tok = xk[n];
        tok = tok < 0 ? 0 : (tok > 95 ? 95 : tok);
        v = (tok == c - 32) ? 1.f : 0.f;
    } else {
        v = xs[n * 2 + (c - 128)];
    }
    h[i] = v;
}

// ---------------- residual init: z = h ----------------
__global__ void gine_copy(const float* __restrict__ a, float* __restrict__ b, int n) {
    int i = blockIdx.x * blockDim.x + threadIdx.x;
    if (i < n) b[i] = a[i];
}

// ---------------- edge gather + relu + scatter-add ----------------
__global__ void gine_edge(const float* __restrict__ h, const int* __restrict__ src,
                          const int* __restrict__ dst, const int* __restrict__ et,
                          float* __restrict__ z) {
    int e = blockIdx.x * blockDim.y + threadIdx.y;
    int c = threadIdx.x;
    if (e >= EE || c >= CC) return;
    int s = src[e], d = dst[e];
    float v = h[s * CC + c] + ((c == et[e]) ? 1.f : 0.f);
    v = fmaxf(v, 0.f);
    if (v != 0.f)
        __hip_atomic_fetch_add(&z[d * CC + c], v, __ATOMIC_RELAXED, __HIP_MEMORY_SCOPE_AGENT);
}

// ---------------- weight prep: f32 [K][N] -> f16 [NPAD][KPAD] (n-major) ----------------
__global__ void gine_prep_weights(const float* __restrict__ W1, const float* __restrict__ W2,
                                  const float* __restrict__ Wh1, const float* __restrict__ Wh2,
                                  _Float16* __restrict__ Wt) {
    int i = blockIdx.x * blockDim.x + threadIdx.x;
    if (i >= 10 * MAT_ELEMS) return;
    int mat = i / MAT_ELEMS;
    int rem = i % MAT_ELEMS;
    int n = rem / KPAD;
    int k = rem % KPAD;
    float v = 0.f;
    if (k < CC && n < CC) {
        if (mat < 4)       v = W1[(size_t)mat * CC * CC + k * CC + n];
        else if (mat < 8)  v = W2[(size_t)(mat - 4) * CC * CC + k * CC + n];
        else if (mat == 8) v = Wh1[k * CC + n];
        else               v = (n < 2) ? Wh2[k * 2 + n] : 0.f;
    }
    Wt[i] = (_Float16)v;
}

// ---------------- fused (z@W1+b1 -> relu) @ W2 + b2 -> [BN] -> [relu] ----------------
union Frag { uint32_t u[8]; v16h h; };

__global__ __launch_bounds__(256) void gine_fused_mlp(
        const float* __restrict__ in,      // [nrows, CC]
        float* __restrict__ out,           // [nrows, out_ld]
        const _Float16* __restrict__ Wt1,  // [NPAD][KPAD]
        const _Float16* __restrict__ Wt2,  // [NPAD][KPAD]
        const float* __restrict__ b1,      // [CC]
        const float* __restrict__ b2,      // [out_cols]
        const float* __restrict__ gamma, const float* __restrict__ beta,
        const float* __restrict__ rmean, const float* __restrict__ rvar,
        int nrows, int use_bn, int relu_out, int out_cols, int out_ld) {
    __shared__ _Float16 tile[WPB][16 * KPAD];

    const int wave = threadIdx.x >> 5;
    const int lane = threadIdx.x & 31;
    const int row_base = (blockIdx.x * WPB + wave) * 16;
    _Float16* T = tile[wave];

    // stage A tile: [16][KPAD] f16, zero-padded
    for (int r = 0; r < 16; ++r) {
        int node = row_base + r;
        for (int col = lane; col < KPAD; col += 32) {
            float v = 0.f;
            if (node < nrows && col < CC) v = in[node * CC + col];
            T[r * KPAD + col] = (_Float16)v;
        }
    }
    __syncthreads();

    const int nl = lane & 15;   // column within 16-wide N tile / row m for A
    const int kh = lane >> 4;   // K half selector
    const uint32_t* T32 = (const uint32_t*)T;
    const int rowdw = nl * (KPAD / 2);   // A row offset in dwords

    v8f acc[NT];
#pragma unroll
    for (int t = 0; t < NT; ++t) acc[t] = (v8f){};

    // ---- GEMM 1: z @ W1 ----
#pragma unroll
    for (int ks = 0; ks < KSTEPS; ++ks) {
        const int kb2 = (ks * 32) >> 1;  // k base in dwords
        Frag a;
#pragma unroll
        for (int j = 0; j < 4; ++j) {
            a.u[j]     = T32[rowdw + kb2 + kh * 4 + j];
            a.u[4 + j] = T32[rowdw + kb2 + 8 + kh * 4 + j];
        }
#pragma unroll
        for (int t = 0; t < NT; ++t) {
            const uint32_t* w32 = (const uint32_t*)(Wt1 + (t * 16 + nl) * KPAD);
            Frag b;
#pragma unroll
            for (int j = 0; j < 8; ++j) b.u[j] = w32[kb2 + (j >> 2) * 8 + kh * 4 + (j & 3)];
            acc[t] = __builtin_amdgcn_wmma_f32_16x16x32_f16(
                false, a.h, false, b.h, (short)0, acc[t], false, false);
        }
    }

    __syncthreads();
    // epilogue 1: +b1, relu, back into LDS tile as f16 A for GEMM 2
#pragma unroll
    for (int t = 0; t < NT; ++t) {
#pragma unroll
        for (int r = 0; r < 8; ++r) {
            int m = r + kh * 8;
            int col = t * 16 + nl;
            float v = acc[t][r] + ((col < CC) ? b1[col] : 0.f);
            v = fmaxf(v, 0.f);
            T[m * KPAD + col] = (_Float16)v;
        }
    }
    // zero K padding cols [NPAD, KPAD)
    for (int idx = lane; idx < 16 * (KPAD - NPAD); idx += 32) {
        int m = idx / (KPAD - NPAD), c2 = NPAD + idx % (KPAD - NPAD);
        T[m * KPAD + c2] = (_Float16)0.f;
    }
    __syncthreads();

    // ---- GEMM 2: t @ W2 ----
#pragma unroll
    for (int t = 0; t < NT; ++t) acc[t] = (v8f){};
#pragma unroll
    for (int ks = 0; ks < KSTEPS; ++ks) {
        const int kb2 = (ks * 32) >> 1;
        Frag a;
#pragma unroll
        for (int j = 0; j < 4; ++j) {
            a.u[j]     = T32[rowdw + kb2 + kh * 4 + j];
            a.u[4 + j] = T32[rowdw + kb2 + 8 + kh * 4 + j];
        }
#pragma unroll
        for (int t = 0; t < NT; ++t) {
            const uint32_t* w32 = (const uint32_t*)(Wt2 + (t * 16 + nl) * KPAD);
            Frag b;
#pragma unroll
            for (int j = 0; j < 8; ++j) b.u[j] = w32[kb2 + (j >> 2) * 8 + kh * 4 + (j & 3)];
            acc[t] = __builtin_amdgcn_wmma_f32_16x16x32_f16(
                false, a.h, false, b.h, (short)0, acc[t], false, false);
        }
    }

    // epilogue 2: +b2, optional BN (eval), optional relu, store f32
#pragma unroll
    for (int t = 0; t < NT; ++t) {
#pragma unroll
        for (int r = 0; r < 8; ++r) {
            int m = r + kh * 8;
            int col = t * 16 + nl;
            int node = row_base + m;
            if (node < nrows && col < out_cols) {
                float v = acc[t][r] + b2[col];
                if (use_bn)
                    v = (v - rmean[col]) * rsqrtf(rvar[col] + 1e-5f) * gamma[col] + beta[col];
                if (relu_out) v = fmaxf(v, 0.f);
                out[node * out_ld + col] = v;
            }
        }
    }
}

// ---------------- host-side orchestration ----------------
extern "C" void kernel_launch(void* const* d_in, const int* in_sizes, int n_in,
                              void* d_out, int out_size, void* d_ws, size_t ws_size,
                              hipStream_t stream) {
    (void)in_sizes; (void)n_in; (void)out_size; (void)ws_size;

    const int*   x_type = (const int*)d_in[0];
    const int*   x_tok  = (const int*)d_in[1];
    const float* x_small= (const float*)d_in[2];
    const int*   eidx   = (const int*)d_in[3];
    const int*   etype  = (const int*)d_in[4];
    // d_in[5] = batch (unused by reference output path)
    const float* W1     = (const float*)d_in[6];
    const float* b1     = (const float*)d_in[7];
    const float* W2     = (const float*)d_in[8];
    const float* b2     = (const float*)d_in[9];
    const float* gamma  = (const float*)d_in[10];
    const float* beta   = (const float*)d_in[11];
    const float* rmean  = (const float*)d_in[12];
    const float* rvar   = (const float*)d_in[13];
    const float* Wh1    = (const float*)d_in[14];
    const float* bh1    = (const float*)d_in[15];
    const float* Wh2    = (const float*)d_in[16];
    const float* bh2    = (const float*)d_in[17];

    const int* src = eidx;
    const int* dst = eidx + EE;

    char* ws = (char*)d_ws;
    const size_t szH = (size_t)NN * CC * sizeof(float);   // 26 MB, 16B-aligned
    float*    hA = (float*)(ws);
    float*    hB = (float*)(ws + szH);
    float*    zb = (float*)(ws + 2 * szH);
    _Float16* Wt = (_Float16*)(ws + 3 * szH);             // 10 * 144*160 f16

    const int elemNC = NN * CC;
    const int gemmGrid = (NN + ROWS_PER_BLOCK - 1) / ROWS_PER_BLOCK;

    gine_prep_weights<<<(10 * MAT_ELEMS + 255) / 256, 256, 0, stream>>>(W1, W2, Wh1, Wh2, Wt);
    gine_build_features<<<(elemNC + 255) / 256, 256, 0, stream>>>(x_type, x_tok, x_small, hA);

    float* hc = hA;
    float* hn = hB;
    for (int l = 0; l < LL; ++l) {
        gine_copy<<<(elemNC + 255) / 256, 256, 0, stream>>>(hc, zb, elemNC);
        gine_edge<<<dim3(EE / 2), dim3(160, 2), 0, stream>>>(hc, src, dst, etype, zb);
        gine_fused_mlp<<<gemmGrid, 256, 0, stream>>>(
            zb, hn,
            Wt + (size_t)l * MAT_ELEMS, Wt + (size_t)(4 + l) * MAT_ELEMS,
            b1 + l * CC, b2 + l * CC,
            gamma + l * CC, beta + l * CC, rmean + l * CC, rvar + l * CC,
            NN, /*use_bn=*/1, /*relu_out=*/1, /*out_cols=*/CC, /*out_ld=*/CC);
        float* tmp = hc; hc = hn; hn = tmp;
    }

    // head: relu(h @ Wh1 + bh1) @ Wh2 + bh2  -> [N, 2]
    gine_fused_mlp<<<gemmGrid, 256, 0, stream>>>(
        hc, (float*)d_out,
        Wt + (size_t)8 * MAT_ELEMS, Wt + (size_t)9 * MAT_ELEMS,
        bh1, bh2,
        nullptr, nullptr, nullptr, nullptr,
        NN, /*use_bn=*/0, /*relu_out=*/0, /*out_cols=*/2, /*out_ld=*/2);
}
